// TransformerEncoderLayer_53798760350072
// MI455X (gfx1250) — compile-verified, matrix operand order checked by reading, not run on hardware
//
#include <hip/hip_runtime.h>
#include <hip/hip_bf16.h>

#define DIM   512
#define SEQ   2048
#define BATCH 4
#define HEADS 8
#define HD    64
#define FFD   2048
#define MROWS 8192   // BATCH*SEQ

typedef __attribute__((ext_vector_type(16))) __bf16 v16bf;
typedef __attribute__((ext_vector_type(8)))  __bf16 v8bf;
typedef __attribute__((ext_vector_type(8)))  float  v8f;

__device__ __forceinline__ unsigned short f2bf(float f) {
  unsigned int u = __float_as_uint(f);
  u += 0x7FFFu + ((u >> 16) & 1u);          // round-to-nearest-even
  return (unsigned short)(u >> 16);
}
__device__ __forceinline__ float bf2f(unsigned short h) {
  return __uint_as_float(((unsigned int)h) << 16);
}
// Build a 16-element bf16 fragment from two aligned 16B chunks at p and p+16
// (matches 16-bit A/B operand layout: K groups 0..15 and 16..31, lane/16 sub-select).
__device__ __forceinline__ v16bf frag16(const unsigned short* p) {
  v8bf lo = *(const v8bf*)p;
  v8bf hi = *(const v8bf*)(p + 16);
  return __builtin_shufflevector(lo, hi, 0,1,2,3,4,5,6,7,8,9,10,11,12,13,14,15);
}
__device__ __forceinline__ v8f wmma_bf16(v16bf a, v16bf b, v8f c) {
  return __builtin_amdgcn_wmma_f32_16x16x32_bf16(false, a, false, b, (short)0, c, false, false);
}

// ---- gfx1250 async global->LDS copy (ASYNCcnt path), with safe fallback ----
#if defined(__has_builtin)
#if __has_builtin(__builtin_amdgcn_global_load_async_to_lds_b128)
#define ASYNC_LDS 1
#endif
#endif

#ifdef ASYNC_LDS
typedef int v4i_vs __attribute__((vector_size(16)));
typedef __attribute__((address_space(1))) v4i_vs* gptr_b128;
typedef __attribute__((address_space(3))) v4i_vs* lptr_b128;
__device__ __forceinline__ void async_copy16(const unsigned short* g, unsigned short* l) {
  __builtin_amdgcn_global_load_async_to_lds_b128((gptr_b128)g, (lptr_b128)l, 0, 0);
}
__device__ __forceinline__ void wait_async() {
#if __has_builtin(__builtin_amdgcn_s_wait_asynccnt)
  __builtin_amdgcn_s_wait_asynccnt(0);
#else
  asm volatile("s_wait_asynccnt 0" ::: "memory");
#endif
}
#else
__device__ __forceinline__ void async_copy16(const unsigned short* g, unsigned short* l) {
  uint4 v = *(const uint4*)g;
  *(uint4*)l = v;
}
__device__ __forceinline__ void wait_async() {}
#endif

// ---------------- LayerNorm + f32 -> bf16 ----------------
__global__ void ln_bf16_kernel(const float* __restrict__ x, const float* __restrict__ g,
                               const float* __restrict__ bb, unsigned short* __restrict__ out) {
  int row = blockIdx.x;
  int t = threadIdx.x;                       // 128 threads, 4 floats each
  const float4* xr = (const float4*)(x + (size_t)row * DIM);
  float4 v = xr[t];
  float s = v.x + v.y + v.z + v.w;
  float q = v.x*v.x + v.y*v.y + v.z*v.z + v.w*v.w;
  for (int o = 16; o > 0; o >>= 1) { s += __shfl_xor(s, o, 32); q += __shfl_xor(q, o, 32); }
  __shared__ float ss[4], qq[4];
  int w = t >> 5;
  if ((t & 31) == 0) { ss[w] = s; qq[w] = q; }
  __syncthreads();
  s = ss[0] + ss[1] + ss[2] + ss[3];
  q = qq[0] + qq[1] + qq[2] + qq[3];
  float mu  = s * (1.0f / DIM);
  float inv = rsqrtf(q * (1.0f / DIM) - mu * mu + 1e-5f);
  float4 gv = ((const float4*)g)[t];
  float4 bv = ((const float4*)bb)[t];
  float o0 = (v.x - mu) * inv * gv.x + bv.x;
  float o1 = (v.y - mu) * inv * gv.y + bv.y;
  float o2 = (v.z - mu) * inv * gv.z + bv.z;
  float o3 = (v.w - mu) * inv * gv.w + bv.w;
  unsigned long long pk = (unsigned long long)f2bf(o0)
      | ((unsigned long long)f2bf(o1) << 16)
      | ((unsigned long long)f2bf(o2) << 32)
      | ((unsigned long long)f2bf(o3) << 48);
  ((unsigned long long*)(out + (size_t)row * DIM))[t] = pk;
}

// ---------------- f32 -> bf16 convert (weights) ----------------
__global__ void cvt_bf16_kernel(const float* __restrict__ in, unsigned short* __restrict__ out, int n) {
  int i = blockIdx.x * blockDim.x + threadIdx.x;
  int stride = gridDim.x * blockDim.x;
  for (; i < n; i += stride) out[i] = f2bf(in[i]);
}

// ---------------- bf16 WMMA GEMM: C(M,N) = A(M,K) @ B(K,N) ----------------
// Block tile 128x128, 8 waves (4x2), each wave 32x64 (2x4 WMMA accum tiles).
// EPI 0: out bf16 (no bias). EPI 1: bias + exact GELU -> bf16. EPI 2: bias + residual -> f32.
template<int EPI>
__global__ void gemm_bf16_kernel(const unsigned short* __restrict__ A,
                                 const unsigned short* __restrict__ Bm,
                                 int N, int K,
                                 const float* __restrict__ bias,
                                 const float* __restrict__ resid,
                                 float* __restrict__ outF,
                                 unsigned short* __restrict__ outBf) {
  __shared__ unsigned short As[128 * 48];   // 128 x 32 tile, stride 48 (16B-aligned rows)
  __shared__ unsigned short Bt[128 * 48];   // B tile transposed: [n][k]
  int tid = threadIdx.x;                    // 256 threads = 8 waves
  int bn = blockIdx.x * 128;
  int bm = blockIdx.y * 128;
  int lane = tid & 31, wave = tid >> 5;
  int wm = wave & 3, wn = wave >> 2;        // 4x2 wave grid -> 128x128 tile
  int lr = lane & 15, lh = lane >> 4;

  int rowA  = tid >> 1;
  int k0    = (tid & 1) * 16;
  int kRowB = tid >> 3;
  int n0    = (tid & 7) * 16;

  v8f z = {0.f,0.f,0.f,0.f,0.f,0.f,0.f,0.f};
  v8f acc[2][4] = {{z, z, z, z}, {z, z, z, z}};

  for (int kt = 0; kt < K; kt += 32) {
    // A tile: straight copy -> async global->LDS (b128 per lane)
    {
      const unsigned short* ga = A + (size_t)(bm + rowA) * K + kt + k0;
      unsigned short* sa = &As[rowA * 48 + k0];
      async_copy16(ga, sa);
      async_copy16(ga + 8, sa + 8);
    }
    // B tile: staged transposed (manual scatter)
    {
      const uint4* gb = (const uint4*)(Bm + (size_t)(kt + kRowB) * N + bn + n0);
      uint4 b0 = gb[0], b1 = gb[1];
      unsigned short tmp[16];
      *(uint4*)(tmp + 0) = b0;
      *(uint4*)(tmp + 8) = b1;
      #pragma unroll
      for (int i = 0; i < 16; ++i) Bt[(n0 + i) * 48 + kRowB] = tmp[i];
    }
    if (kt + 32 < K) __builtin_prefetch(Bm + (size_t)(kt + 32 + kRowB) * N + bn + n0);
    wait_async();
    __syncthreads();

    v16bf fa[2], fb[4];
    #pragma unroll
    for (int mt = 0; mt < 2; ++mt)
      fa[mt] = frag16(&As[(wm * 32 + mt * 16 + lr) * 48 + lh * 8]);
    #pragma unroll
    for (int nt = 0; nt < 4; ++nt)
      fb[nt] = frag16(&Bt[(wn * 64 + nt * 16 + lr) * 48 + lh * 8]);
    #pragma unroll
    for (int mt = 0; mt < 2; ++mt)
      #pragma unroll
      for (int nt = 0; nt < 4; ++nt)
        acc[mt][nt] = wmma_bf16(fa[mt], fb[nt], acc[mt][nt]);
    __syncthreads();
  }

  #pragma unroll
  for (int mt = 0; mt < 2; ++mt)
    #pragma unroll
    for (int nt = 0; nt < 4; ++nt) {
      int gc  = bn + wn * 64 + nt * 16 + lr;
      int gr0 = bm + wm * 32 + mt * 16 + lh * 8;
      #pragma unroll
      for (int r = 0; r < 8; ++r) {
        float v = acc[mt][nt][r];
        size_t idx = (size_t)(gr0 + r) * (size_t)N + gc;
        if (EPI == 0) {
          outBf[idx] = f2bf(v);
        } else if (EPI == 1) {
          v += bias[gc];
          v = 0.5f * v * (1.0f + erff(v * 0.70710678118f));   // exact GELU
          outBf[idx] = f2bf(v);
        } else {
          v += bias[gc] + resid[idx];
          outF[idx] = v;
        }
      }
    }
}

// ---------------- flash attention (one block = 64 q rows of one (b,h)) ----------------
__global__ void attn_kernel(const unsigned short* __restrict__ qkv,
                            unsigned short* __restrict__ outBf) {
  __shared__ unsigned short Qs[64 * 80];
  __shared__ unsigned short Ks[64 * 80];
  __shared__ unsigned short Vt[64 * 80];       // transposed: [d][kv]
  __shared__ unsigned short Ps[4 * 16 * 80];   // per-wave P strip
  int qt = blockIdx.x;                          // query tile 0..31
  int bh = blockIdx.y;                          // 0..31
  int bb = bh >> 3, hh = bh & 7;
  int tid = threadIdx.x;                        // 128 threads = 4 waves
  int wave = tid >> 5, lane = tid & 31, lr = lane & 15, lh = lane >> 4;
  size_t rowbase = (size_t)bb * SEQ;
  int r2 = tid >> 1, half = tid & 1;

  { // load Q tile (64x64 bf16) via async copy
    const unsigned short* g = qkv + (rowbase + qt * 64 + r2) * 1536 + hh * 64 + half * 32;
    unsigned short* s = &Qs[r2 * 80 + half * 32];
    async_copy16(g, s);      async_copy16(g + 8, s + 8);
    async_copy16(g + 16, s + 16); async_copy16(g + 24, s + 24);
  }
  wait_async();
  __syncthreads();
  v16bf fqs[2];
  fqs[0] = frag16(&Qs[(wave * 16 + lr) * 80 +  0 + lh * 8]);
  fqs[1] = frag16(&Qs[(wave * 16 + lr) * 80 + 32 + lh * 8]);

  float mrow[8], lrow[8];
  #pragma unroll
  for (int r = 0; r < 8; ++r) { mrow[r] = -1e30f; lrow[r] = 0.0f; }
  v8f z = {0.f,0.f,0.f,0.f,0.f,0.f,0.f,0.f};
  v8f oacc[4] = {z, z, z, z};
  unsigned short* pp = &Ps[wave * 16 * 80];

  for (int kt = 0; kt < 32; ++kt) {
    { // K tile row-major via async copy
      const unsigned short* g = qkv + (rowbase + kt * 64 + r2) * 1536 + 512 + hh * 64 + half * 32;
      unsigned short* s = &Ks[r2 * 80 + half * 32];
      async_copy16(g, s);      async_copy16(g + 8, s + 8);
      async_copy16(g + 16, s + 16); async_copy16(g + 24, s + 24);
    }
    { // V tile transposed into Vt[d][kv]
      const uint4* g = (const uint4*)(qkv + (rowbase + kt * 64 + r2) * 1536 + 1024 + hh * 64 + half * 32);
      uint4 t0 = g[0], t1 = g[1], t2 = g[2], t3 = g[3];
      unsigned short arr[32];
      *(uint4*)(arr +  0) = t0; *(uint4*)(arr +  8) = t1;
      *(uint4*)(arr + 16) = t2; *(uint4*)(arr + 24) = t3;
      #pragma unroll
      for (int i = 0; i < 32; ++i) Vt[(half * 32 + i) * 80 + r2] = arr[i];
    }
    wait_async();
    __syncthreads();

    // S = Q @ K^T  (wave strip: 16 x 64)
    v8f sacc[4] = {z, z, z, z};
    #pragma unroll
    for (int ks = 0; ks < 2; ++ks)
      #pragma unroll
      for (int t = 0; t < 4; ++t) {
        v16bf fk = frag16(&Ks[(t * 16 + lr) * 80 + ks * 32 + lh * 8]);
        sacc[t] = wmma_bf16(fqs[ks], fk, sacc[t]);
      }

    // online softmax update (row = lh*8 + r, reduce over 16 lanes of each half)
    float corr[8];
    #pragma unroll
    for (int r = 0; r < 8; ++r) {
      float a0 = sacc[0][r] * 0.125f, a1 = sacc[1][r] * 0.125f;
      float a2 = sacc[2][r] * 0.125f, a3 = sacc[3][r] * 0.125f;
      float mx = fmaxf(fmaxf(a0, a1), fmaxf(a2, a3));
      for (int off = 8; off > 0; off >>= 1) mx = fmaxf(mx, __shfl_xor(mx, off, 32));
      float mnew = fmaxf(mrow[r], mx);
      float c = __expf(mrow[r] - mnew);
      corr[r] = c; mrow[r] = mnew;
      a0 = __expf(a0 - mnew); a1 = __expf(a1 - mnew);
      a2 = __expf(a2 - mnew); a3 = __expf(a3 - mnew);
      sacc[0][r] = a0; sacc[1][r] = a1; sacc[2][r] = a2; sacc[3][r] = a3;
      float sm = a0 + a1 + a2 + a3;
      for (int off = 8; off > 0; off >>= 1) sm += __shfl_xor(sm, off, 32);
      lrow[r] = lrow[r] * c + sm;
    }

    // stash P (C-layout -> LDS), re-read as A fragments
    #pragma unroll
    for (int r = 0; r < 8; ++r) {
      int rowl = lh * 8 + r;
      #pragma unroll
      for (int t = 0; t < 4; ++t) pp[rowl * 80 + t * 16 + lr] = f2bf(sacc[t][r]);
    }
    asm volatile("s_wait_dscnt 0" ::: "memory");

    #pragma unroll
    for (int r = 0; r < 8; ++r) {
      oacc[0][r] *= corr[r]; oacc[1][r] *= corr[r];
      oacc[2][r] *= corr[r]; oacc[3][r] *= corr[r];
    }
    // O += P @ V
    #pragma unroll
    for (int ks = 0; ks < 2; ++ks) {
      v16bf fp = frag16(&pp[lr * 80 + ks * 32 + lh * 8]);
      #pragma unroll
      for (int t = 0; t < 4; ++t) {
        v16bf fv = frag16(&Vt[(t * 16 + lr) * 80 + ks * 32 + lh * 8]);
        oacc[t] = wmma_bf16(fp, fv, oacc[t]);
      }
    }
    __syncthreads();
  }

  #pragma unroll
  for (int r = 0; r < 8; ++r) {
    int rowl = lh * 8 + r;
    size_t gr = rowbase + qt * 64 + wave * 16 + rowl;
    float dn = 1.0f / lrow[r];
    #pragma unroll
    for (int t = 0; t < 4; ++t) {
      int gc = hh * 64 + t * 16 + lr;
      outBf[gr * DIM + gc] = f2bf(oacc[t][r] * dn);
    }
  }
}

// ---------------- q_attn = softmax(q0 . K^T) per (b,h) ----------------
__global__ void qattn_kernel(const unsigned short* __restrict__ qkv, float* __restrict__ out) {
  __shared__ float q0[64];
  __shared__ float red[8];
  int bh = blockIdx.x, bb = bh >> 3, hh = bh & 7;
  int tid = threadIdx.x;                       // 256
  int wave = tid >> 5, lane = tid & 31;
  if (tid < 64) q0[tid] = bf2f(qkv[(size_t)bb * SEQ * 1536 + hh * 64 + tid]);
  __syncthreads();
  float dots[8];
  #pragma unroll
  for (int i = 0; i < 8; ++i) {
    int j = tid + i * 256;
    const unsigned short* kr = qkv + ((size_t)bb * SEQ + j) * 1536 + 512 + hh * 64;
    float d = 0.0f;
    for (int c = 0; c < 64; ++c) d += q0[c] * bf2f(kr[c]);
    dots[i] = d * 0.125f;
  }
  float mx = dots[0];
  #pragma unroll
  for (int i = 1; i < 8; ++i) mx = fmaxf(mx, dots[i]);
  for (int off = 16; off > 0; off >>= 1) mx = fmaxf(mx, __shfl_xor(mx, off, 32));
  if (lane == 0) red[wave] = mx;
  __syncthreads();
  mx = red[0];
  for (int i = 1; i < 8; ++i) mx = fmaxf(mx, red[i]);
  __syncthreads();
  float sm = 0.0f;
  #pragma unroll
  for (int i = 0; i < 8; ++i) { dots[i] = __expf(dots[i] - mx); sm += dots[i]; }
  for (int off = 16; off > 0; off >>= 1) sm += __shfl_xor(sm, off, 32);
  if (lane == 0) red[wave] = sm;
  __syncthreads();
  float tot = 0.0f;
  for (int i = 0; i < 8; ++i) tot += red[i];
  float inv = 1.0f / tot;
  #pragma unroll
  for (int i = 0; i < 8; ++i) out[(size_t)bh * SEQ + tid + i * 256] = dots[i] * inv;
}

// ---------------- host launch ----------------
extern "C" void kernel_launch(void* const* d_in, const int* in_sizes, int n_in,
                              void* d_out, int out_size, void* d_ws, size_t ws_size,
                              hipStream_t stream) {
  (void)in_sizes; (void)n_in; (void)out_size; (void)ws_size;
  const float* x      = (const float*)d_in[0];
  const float* ln1_g  = (const float*)d_in[1];
  const float* ln1_b  = (const float*)d_in[2];
  const float* w_qkv  = (const float*)d_in[3];
  const float* w_proj = (const float*)d_in[4];
  const float* b_proj = (const float*)d_in[5];
  const float* ln2_g  = (const float*)d_in[6];
  const float* ln2_b  = (const float*)d_in[7];
  const float* w1     = (const float*)d_in[8];
  const float* b1     = (const float*)d_in[9];
  const float* w2     = (const float*)d_in[10];
  const float* b2     = (const float*)d_in[11];
  float* outX = (float*)d_out;
  float* outQ = outX + (size_t)MROWS * DIM;

  char* ws = (char*)d_ws;
  size_t off = 0;
  auto take = [&](size_t bytes) { void* p = ws + off; off += bytes; return p; };
  unsigned short* xn_bf    = (unsigned short*)take((size_t)MROWS * DIM * 2);
  unsigned short* wqkv_bf  = (unsigned short*)take((size_t)DIM * 3 * DIM * 2);
  unsigned short* wproj_bf = (unsigned short*)take((size_t)DIM * DIM * 2);
  unsigned short* w1_bf    = (unsigned short*)take((size_t)DIM * FFD * 2);
  unsigned short* w2_bf    = (unsigned short*)take((size_t)FFD * DIM * 2);
  unsigned short* qkv_bf   = (unsigned short*)take((size_t)MROWS * 3 * DIM * 2);
  unsigned short* ao_bf    = (unsigned short*)take((size_t)MROWS * DIM * 2);
  float*          x2       = (float*)take((size_t)MROWS * DIM * 4);
  unsigned short* xn2_bf   = (unsigned short*)take((size_t)MROWS * DIM * 2);
  unsigned short* ff_bf    = (unsigned short*)take((size_t)MROWS * FFD * 2);

  ln_bf16_kernel<<<MROWS, 128, 0, stream>>>(x, ln1_g, ln1_b, xn_bf);
  cvt_bf16_kernel<<<1024, 256, 0, stream>>>(w_qkv, wqkv_bf, DIM * 3 * DIM);
  cvt_bf16_kernel<<<512, 256, 0, stream>>>(w_proj, wproj_bf, DIM * DIM);
  cvt_bf16_kernel<<<1024, 256, 0, stream>>>(w1, w1_bf, DIM * FFD);
  cvt_bf16_kernel<<<1024, 256, 0, stream>>>(w2, w2_bf, FFD * DIM);

  gemm_bf16_kernel<0><<<dim3(3 * DIM / 128, MROWS / 128), 256, 0, stream>>>(
      xn_bf, wqkv_bf, 3 * DIM, DIM, nullptr, nullptr, nullptr, qkv_bf);

  attn_kernel<<<dim3(SEQ / 64, BATCH * HEADS), 128, 0, stream>>>(qkv_bf, ao_bf);
  qattn_kernel<<<BATCH * HEADS, 256, 0, stream>>>(qkv_bf, outQ);

  gemm_bf16_kernel<2><<<dim3(DIM / 128, MROWS / 128), 256, 0, stream>>>(
      ao_bf, wproj_bf, DIM, DIM, b_proj, x, x2, nullptr);

  ln_bf16_kernel<<<MROWS, 128, 0, stream>>>(x2, ln2_g, ln2_b, xn2_bf);

  gemm_bf16_kernel<1><<<dim3(FFD / 128, MROWS / 128), 256, 0, stream>>>(
      xn2_bf, w1_bf, FFD, DIM, b1, nullptr, nullptr, ff_bf);

  gemm_bf16_kernel<2><<<dim3(DIM / 128, MROWS / 128), 256, 0, stream>>>(
      ff_bf, w2_bf, DIM, FFD, b2, x2, outX, nullptr);
}
